// DSA_29497835389579
// MI455X (gfx1250) — matrix operand order repaired and checked by reading, not compiled, not run on hardware
//
#include <hip/hip_runtime.h>
#include <math.h>

#define BB 2
#define SS 2048
#define DD 2048
#define HH 16

typedef __attribute__((ext_vector_type(16))) __bf16 v16bf;
typedef __attribute__((ext_vector_type(8)))  __bf16 v8bf;
typedef __attribute__((ext_vector_type(8)))  float  v8f;

union Frag { v16bf v; v8bf h[2]; };

__device__ __forceinline__ v8f wmma_bf16(const Frag& a, const Frag& b, v8f c){
  return __builtin_amdgcn_wmma_f32_16x16x32_bf16(false, a.v, false, b.v, (short)0, c, false, false);
}

// 32-byte memory -> LDS async copy (two b128 ops; INST_OFFSET applies to both sides)
__device__ __forceinline__ void async_cp16x2(unsigned lds, const __bf16* g){
  unsigned long long ga = (unsigned long long)(uintptr_t)g;
  asm volatile("global_load_async_to_lds_b128 %0, %1, off\n\t"
               "global_load_async_to_lds_b128 %0, %1, off offset:16"
               :: "v"(lds), "v"(ga) : "memory");
}

__device__ __forceinline__ void wait_async0(){
  asm volatile("s_wait_asynccnt 0x0" ::: "memory");
}

// ---------------- elementwise helpers ----------------

__global__ void cast_f32_bf16(const float* __restrict__ src, __bf16* __restrict__ dst, int n){
  int i = blockIdx.x * 256 + threadIdx.x;
  if (i < n) dst[i] = (__bf16)src[i];
}

// W[K,N] (row-major) -> Wt[N,K] (row-major), output-coalesced
__global__ void cast_transpose_f32_bf16(const float* __restrict__ W, __bf16* __restrict__ Wt,
                                        int K, int N){
  int i = blockIdx.x * 256 + threadIdx.x;
  if (i < N * K){
    int n = i / K, k = i - n * K;
    Wt[i] = (__bf16)W[(size_t)k * N + n];
  }
}

__global__ void rmsnorm_bf16(const float* __restrict__ X, const float* __restrict__ w,
                             __bf16* __restrict__ Y, int C){
  __shared__ float red[256];
  const int row = blockIdx.x;
  const float* x = X + (size_t)row * C;
  float s = 0.f;
  for (int c = threadIdx.x; c < C; c += 256){ float v = x[c]; s += v * v; }
  red[threadIdx.x] = s;
  __syncthreads();
  for (int o = 128; o > 0; o >>= 1){
    if (threadIdx.x < o) red[threadIdx.x] += red[threadIdx.x + o];
    __syncthreads();
  }
  const float rs = rsqrtf(red[0] / (float)C + 1e-6f);
  __bf16* y = Y + (size_t)row * C;
  for (int c = threadIdx.x; c < C; c += 256) y[c] = (__bf16)(x[c] * rs * w[c]);
}

// q: [B,H,S,192] bf16 from q_C [B,S,H*128] and rope(q_R [B,S,H*64])
__global__ void assemble_q(const float* __restrict__ qc, const float* __restrict__ qr,
                           const float* __restrict__ cosT, const float* __restrict__ sinT,
                           __bf16* __restrict__ q){
  const int bhs = blockIdx.x;
  const int s  = bhs & (SS - 1);
  const int bh = bhs >> 11;            // SS == 2048
  const int h  = bh & (HH - 1);
  const int b  = bh >> 4;
  const int t  = threadIdx.x;
  __bf16* dst = q + ((size_t)bh * SS + s) * 192;
  if (t < 128){
    dst[t] = (__bf16)qc[((size_t)b * SS + s) * 2048 + h * 128 + t];
  } else {
    const int j = t - 128, pI = j >> 1;
    const float* src = qr + ((size_t)b * SS + s) * 1024 + h * 64;
    const float x1 = src[2 * pI], x2 = src[2 * pI + 1];
    const float c = cosT[s * 32 + pI], sn = sinT[s * 32 + pI];
    dst[128 + j] = (__bf16)((j & 1) ? (x1 * sn + x2 * c) : (x1 * c - x2 * sn));
  }
}

// k: [B,H,S,192] bf16 from k_C [B,S,H*128] and broadcast rope(k_R [B,S,64])
__global__ void assemble_k(const float* __restrict__ kc, const float* __restrict__ kr,
                           const float* __restrict__ cosT, const float* __restrict__ sinT,
                           __bf16* __restrict__ k){
  const int bhs = blockIdx.x;
  const int s  = bhs & (SS - 1);
  const int bh = bhs >> 11;
  const int h  = bh & (HH - 1);
  const int b  = bh >> 4;
  const int t  = threadIdx.x;
  __bf16* dst = k + ((size_t)bh * SS + s) * 192;
  if (t < 128){
    dst[t] = (__bf16)kc[((size_t)b * SS + s) * 2048 + h * 128 + t];
  } else {
    const int j = t - 128, pI = j >> 1;
    const float* src = kr + ((size_t)b * SS + s) * 64;
    const float x1 = src[2 * pI], x2 = src[2 * pI + 1];
    const float c = cosT[s * 32 + pI], sn = sinT[s * 32 + pI];
    dst[128 + j] = (__bf16)((j & 1) ? (x1 * sn + x2 * c) : (x1 * c - x2 * sn));
  }
}

// v_f [B,S,H*128] f32 -> vt [B,H,128,S] bf16
__global__ void transpose_v(const float* __restrict__ v, __bf16* __restrict__ vt){
  int idx = blockIdx.x * 256 + threadIdx.x;          // B*H*S*128 = 8.4M
  if (idx >= BB * HH * SS * 128) return;
  const int d  = idx & 127;
  const int s  = (idx >> 7) & (SS - 1);
  const int bh = idx >> 18;                          // 128*2048 = 2^18
  const int b  = bh >> 4, h = bh & 15;
  vt[((size_t)bh * 128 + d) * SS + s] = (__bf16)v[((size_t)b * SS + s) * 2048 + h * 128 + d];
}

// ---------------- WMMA GEMM: C[M,N] f32 = A[M,K] bf16 * Bt[N,K]^T bf16 ----------------
// Block tile 128x128, 8 waves (2x4), each wave 64x32 (4x2 of 16x16), K-step 32.
// Double-buffered LDS fed by GLOBAL_LOAD_ASYNC_TO_LDS_B128 (ASYNCcnt pipeline):
// one s_wait_asynccnt + one barrier per K-step; next tile's copies overlap compute.

__global__ void __launch_bounds__(256) gemm_bf16(const __bf16* __restrict__ A,
                                                 const __bf16* __restrict__ Bt,
                                                 float* __restrict__ C,
                                                 int M, int N, int K){
  __shared__ __align__(16) __bf16 As[2][128 * 40];
  __shared__ __align__(16) __bf16 Bs[2][128 * 40];
  const int tid  = threadIdx.x;
  const int lane = tid & 31, w = tid >> 5;
  const int half = lane >> 4, lm = lane & 15;
  const int wm = (w >> 2) * 64;   // wave M offset inside block tile
  const int wn = (w & 3) * 32;    // wave N offset inside block tile
  const int mblk = blockIdx.y * 128, nblk = blockIdx.x * 128;

  v8f acc[4][2];
  for (int mt = 0; mt < 4; mt++)
    for (int nt = 0; nt < 2; nt++)
      for (int e = 0; e < 8; e++) acc[mt][nt][e] = 0.f;

  // per-thread 32B staging slice
  const int lrow = tid >> 1, lcol = (tid & 1) * 16;
  const __bf16* aRow = A + (size_t)(mblk + lrow) * K + lcol;
  const int nrow = nblk + lrow;
  const __bf16* bRow = Bt + (size_t)nrow * K + lcol;
  const bool bOK = (nrow < N);   // stale LDS rows for n>=N only feed discarded columns
  const unsigned ldsA0 = (unsigned)(uintptr_t)&As[0][lrow * 40 + lcol];
  const unsigned ldsA1 = (unsigned)(uintptr_t)&As[1][lrow * 40 + lcol];
  const unsigned ldsB0 = (unsigned)(uintptr_t)&Bs[0][lrow * 40 + lcol];
  const unsigned ldsB1 = (unsigned)(uintptr_t)&Bs[1][lrow * 40 + lcol];

  const int ntiles = K >> 5;
  // prologue: stage tile 0 into buffer 0
  async_cp16x2(ldsA0, aRow);
  if (bOK) async_cp16x2(ldsB0, bRow);

  int buf = 0;
  for (int i = 0; i < ntiles; i++){
    wait_async0();        // this wave's copies into buf have landed
    __syncthreads();      // => all waves' copies landed; all done reading buf^1
    if (i + 1 < ntiles){  // overlap next tile's fetch with this tile's WMMAs
      const int ko = (i + 1) << 5;
      async_cp16x2(buf ? ldsA0 : ldsA1, aRow + ko);
      if (bOK) async_cp16x2(buf ? ldsB0 : ldsB1, bRow + ko);
    }

    const __bf16* as = &As[buf][0];
    const __bf16* bs = &Bs[buf][0];
    Frag af[4], bfr[2];
    for (int mt = 0; mt < 4; mt++){
      const int r = wm + mt * 16 + lm;
      af[mt].h[0] = *(const v8bf*)&as[r * 40 + half * 8];
      af[mt].h[1] = *(const v8bf*)&as[r * 40 + 16 + half * 8];
    }
    for (int nt = 0; nt < 2; nt++){
      const int r = wn + nt * 16 + lm;
      bfr[nt].h[0] = *(const v8bf*)&bs[r * 40 + half * 8];
      bfr[nt].h[1] = *(const v8bf*)&bs[r * 40 + 16 + half * 8];
    }
    for (int mt = 0; mt < 4; mt++)
      for (int nt = 0; nt < 2; nt++)
        acc[mt][nt] = wmma_bf16(af[mt], bfr[nt], acc[mt][nt]);

    buf ^= 1;
  }

  for (int mt = 0; mt < 4; mt++)
    for (int nt = 0; nt < 2; nt++){
      const int ncol = nblk + wn + nt * 16 + lm;
      if (ncol >= N) continue;
      for (int r = 0; r < 8; r++){
        const int mrow = mblk + wm + mt * 16 + r + half * 8;
        if (mrow < M) C[(size_t)mrow * N + ncol] = acc[mt][nt][r];
      }
    }
}

// ---------------- flash attention ----------------
// Q,K: [B,H,S,192] bf16 ; Vt: [B,H,128,S] bf16 ; O: [B,S,H*128] bf16
// grid (S/128, H, B), block 256; wave w handles 16 query rows.

__global__ void __launch_bounds__(256) mla_attn(const __bf16* __restrict__ Q,
                                                const __bf16* __restrict__ Kt,
                                                const __bf16* __restrict__ Vt,
                                                __bf16* __restrict__ O){
  __shared__ __align__(16) __bf16 Plds[8][16 * 40];
  const int tid = threadIdx.x, lane = tid & 31, w = tid >> 5;
  const int half = lane >> 4, lm = lane & 15;
  const size_t bh = (size_t)blockIdx.z * HH + blockIdx.y;
  const int b = blockIdx.z, h = blockIdx.y;
  const int qw = blockIdx.x * 128 + w * 16;
  const float SC = 13.856406460551018f;   // sqrt(192): reference divides by d^-1/2

  // persistent Q fragments (6 x 16x32 bf16)
  Frag qf[6];
  {
    const __bf16* qptr = Q + (bh * SS + qw) * 192 + (size_t)lm * 192;
    for (int dk = 0; dk < 6; dk++){
      qf[dk].h[0] = *(const v8bf*)(qptr + dk * 32 + half * 8);
      qf[dk].h[1] = *(const v8bf*)(qptr + dk * 32 + 16 + half * 8);
    }
  }

  v8f o[8];
  for (int dt = 0; dt < 8; dt++)
    for (int e = 0; e < 8; e++) o[dt][e] = 0.f;
  float mI[8], lI[8];
  for (int r = 0; r < 8; r++){ mI[r] = -__builtin_inff(); lI[r] = 0.f; }

  __bf16* pl = &Plds[w][0];
  const int kend = (qw + 16 < SS) ? (qw + 16) : SS;   // causal: keys <= max query row

  for (int k0 = 0; k0 < kend; k0 += 32){
    const bool tile2 = (k0 + 16) < kend;
    v8f sA, sB;
    for (int e = 0; e < 8; e++){ sA[e] = 0.f; sB[e] = 0.f; }

    // S = Q * K^T  (keys as transposed-B rows)
    for (int dk = 0; dk < 6; dk++){
      Frag kf;
      const __bf16* kp = Kt + (bh * SS + k0 + lm) * 192 + dk * 32 + half * 8;
      kf.h[0] = *(const v8bf*)kp;
      kf.h[1] = *(const v8bf*)(kp + 16);
      sA = wmma_bf16(qf[dk], kf, sA);
    }
    if (tile2){
      for (int dk = 0; dk < 6; dk++){
        Frag kf;
        const __bf16* kp = Kt + (bh * SS + k0 + 16 + lm) * 192 + dk * 32 + half * 8;
        kf.h[0] = *(const v8bf*)kp;
        kf.h[1] = *(const v8bf*)(kp + 16);
        sB = wmma_bf16(qf[dk], kf, sB);
      }
    }

    // online softmax update (row stats reduced over 16 lanes of each half)
    float alpha[8];
    for (int r = 0; r < 8; r++){
      const int q = qw + r + half * 8;
      float a = sA[r] * SC;
      if (k0 + lm > q) a = -__builtin_inff();
      float bb = -__builtin_inff();
      if (tile2){
        bb = sB[r] * SC;
        if (k0 + 16 + lm > q) bb = -__builtin_inff();
      }
      float v0 = fmaxf(a, bb);
      v0 = fmaxf(v0, __shfl_xor(v0, 1, 32));
      v0 = fmaxf(v0, __shfl_xor(v0, 2, 32));
      v0 = fmaxf(v0, __shfl_xor(v0, 4, 32));
      v0 = fmaxf(v0, __shfl_xor(v0, 8, 32));
      const float mn = fmaxf(mI[r], v0);
      alpha[r] = __expf(mI[r] - mn);
      const float pa = __expf(a - mn);
      const float pb = tile2 ? __expf(bb - mn) : 0.f;
      mI[r] = mn;
      float rs = pa + pb;
      rs += __shfl_xor(rs, 1, 32);
      rs += __shfl_xor(rs, 2, 32);
      rs += __shfl_xor(rs, 4, 32);
      rs += __shfl_xor(rs, 8, 32);
      lI[r] = lI[r] * alpha[r] + rs;
      // C-layout -> LDS (row m = r + half*8), cols = local key index
      pl[(r + half * 8) * 40 + lm]      = (__bf16)pa;
      pl[(r + half * 8) * 40 + 16 + lm] = (__bf16)pb;
    }
    for (int dt = 0; dt < 8; dt++)
      for (int r = 0; r < 8; r++) o[dt][r] *= alpha[r];

    // P as A-fragment (16x32) from LDS
    Frag pf;
    {
      const __bf16* pp = pl + lm * 40;
      pf.h[0] = *(const v8bf*)(pp + half * 16);
      pf.h[1] = *(const v8bf*)(pp + half * 16 + 8);
    }
    // V^T rows are output-dims; contiguous keys per lane
    int kb = k0 + half * 16;
    if (kb + 16 > SS) kb = SS - 16;   // OOB-safe; matching P entries are zero
    const __bf16* vbase = Vt + (bh * 128) * SS;
    for (int dt = 0; dt < 8; dt++){
      Frag vf;
      const __bf16* vp = vbase + (size_t)(dt * 16 + lm) * SS + kb;
      vf.h[0] = *(const v8bf*)vp;
      vf.h[1] = *(const v8bf*)(vp + 8);
      o[dt] = wmma_bf16(pf, vf, o[dt]);
    }
  }

  // epilogue: normalize and write [B,S,H*128] bf16
  for (int r = 0; r < 8; r++){
    const float inv = 1.0f / lI[r];
    const int q = qw + r + half * 8;
    __bf16* op = O + ((size_t)b * SS + q) * 2048 + h * 128;
    for (int dt = 0; dt < 8; dt++) op[dt * 16 + lm] = (__bf16)(o[dt][r] * inv);
  }
}

// ---------------- launcher ----------------

extern "C" void kernel_launch(void* const* d_in, const int* in_sizes, int n_in,
                              void* d_out, int out_size, void* d_ws, size_t ws_size,
                              hipStream_t stream){
  (void)in_sizes; (void)n_in; (void)out_size; (void)ws_size;
  const float* x     = (const float*)d_in[0];
  // d_in[1] = mask (causal handled analytically)
  const float* cosT  = (const float*)d_in[2];
  const float* sinT  = (const float*)d_in[3];
  const float* W_DQ  = (const float*)d_in[4];
  const float* W_UQ  = (const float*)d_in[5];
  const float* W_UQR = (const float*)d_in[6];
  const float* W_DKV = (const float*)d_in[7];
  const float* W_UK  = (const float*)d_in[8];
  const float* W_UV  = (const float*)d_in[9];
  const float* W_DKR = (const float*)d_in[10];
  const float* W_O   = (const float*)d_in[11];
  const float* qnw   = (const float*)d_in[12];
  const float* kvnw  = (const float*)d_in[13];
  float* out = (float*)d_out;

  char* p = (char*)d_ws;
  auto alloc = [&](size_t nbytes) -> char* {
    char* r = p; p += (nbytes + 255) & ~(size_t)255; return r;
  };
  const size_t M = (size_t)BB * SS;   // 4096 rows

  __bf16* xb    = (__bf16*)alloc(M * 2048 * 2);
  __bf16* wdqt  = (__bf16*)alloc((size_t)1536 * 2048 * 2);
  __bf16* wuqt  = (__bf16*)alloc((size_t)2048 * 1536 * 2);
  __bf16* wuqrt = (__bf16*)alloc((size_t)1024 * 1536 * 2);
  __bf16* wdkvt = (__bf16*)alloc((size_t)512 * 2048 * 2);
  __bf16* wukt  = (__bf16*)alloc((size_t)2048 * 512 * 2);
  __bf16* wuvt  = (__bf16*)alloc((size_t)2048 * 512 * 2);
  __bf16* wdkrt = (__bf16*)alloc((size_t)64 * 2048 * 2);
  __bf16* wot   = (__bf16*)alloc((size_t)2048 * 2048 * 2);
  float*  cq_f  = (float*)alloc(M * 1536 * 4);
  float*  ckv_f = (float*)alloc(M * 512 * 4);
  float*  kr_f  = (float*)alloc(M * 64 * 4);
  __bf16* cqb   = (__bf16*)alloc(M * 1536 * 2);
  __bf16* ckvb  = (__bf16*)alloc(M * 512 * 2);
  float*  qc_f  = (float*)alloc(M * 2048 * 4);
  float*  qr_f  = (float*)alloc(M * 1024 * 4);
  float*  kc_f  = (float*)alloc(M * 2048 * 4);
  float*  v_f   = (float*)alloc(M * 2048 * 4);
  __bf16* qb    = (__bf16*)alloc((size_t)BB * HH * SS * 192 * 2);
  __bf16* kb    = (__bf16*)alloc((size_t)BB * HH * SS * 192 * 2);
  __bf16* vtb   = (__bf16*)alloc((size_t)BB * HH * 128 * SS * 2);
  __bf16* aob   = (__bf16*)alloc(M * 2048 * 2);
  (void)alloc(4096);   // tail slack for clamped reads

  auto cdiv = [](size_t a, size_t b) -> unsigned { return (unsigned)((a + b - 1) / b); };

  // casts / transposed weight casts
  cast_f32_bf16<<<cdiv(M * 2048, 256), 256, 0, stream>>>(x, xb, (int)(M * 2048));
  cast_transpose_f32_bf16<<<cdiv((size_t)2048 * 1536, 256), 256, 0, stream>>>(W_DQ,  wdqt,  2048, 1536);
  cast_transpose_f32_bf16<<<cdiv((size_t)1536 * 2048, 256), 256, 0, stream>>>(W_UQ,  wuqt,  1536, 2048);
  cast_transpose_f32_bf16<<<cdiv((size_t)1536 * 1024, 256), 256, 0, stream>>>(W_UQR, wuqrt, 1536, 1024);
  cast_transpose_f32_bf16<<<cdiv((size_t)2048 * 512, 256), 256, 0, stream>>>(W_DKV, wdkvt, 2048, 512);
  cast_transpose_f32_bf16<<<cdiv((size_t)512 * 2048, 256), 256, 0, stream>>>(W_UK,  wukt,  512, 2048);
  cast_transpose_f32_bf16<<<cdiv((size_t)512 * 2048, 256), 256, 0, stream>>>(W_UV,  wuvt,  512, 2048);
  cast_transpose_f32_bf16<<<cdiv((size_t)2048 * 64, 256), 256, 0, stream>>>(W_DKR, wdkrt, 2048, 64);
  cast_transpose_f32_bf16<<<cdiv((size_t)2048 * 2048, 256), 256, 0, stream>>>(W_O, wot, 2048, 2048);

  // down projections (+ rope key projection)
  gemm_bf16<<<dim3(12, 32), 256, 0, stream>>>(xb, wdqt,  cq_f,  4096, 1536, 2048);
  gemm_bf16<<<dim3(4, 32),  256, 0, stream>>>(xb, wdkvt, ckv_f, 4096, 512,  2048);
  gemm_bf16<<<dim3(1, 32),  256, 0, stream>>>(xb, wdkrt, kr_f,  4096, 64,   2048);

  // rmsnorm -> bf16 activations
  rmsnorm_bf16<<<4096, 256, 0, stream>>>(cq_f,  qnw,  cqb,  1536);
  rmsnorm_bf16<<<4096, 256, 0, stream>>>(ckv_f, kvnw, ckvb, 512);

  // up projections
  gemm_bf16<<<dim3(16, 32), 256, 0, stream>>>(cqb,  wuqt,  qc_f, 4096, 2048, 1536);
  gemm_bf16<<<dim3(8, 32),  256, 0, stream>>>(cqb,  wuqrt, qr_f, 4096, 1024, 1536);
  gemm_bf16<<<dim3(16, 32), 256, 0, stream>>>(ckvb, wukt,  kc_f, 4096, 2048, 512);
  gemm_bf16<<<dim3(16, 32), 256, 0, stream>>>(ckvb, wuvt,  v_f,  4096, 2048, 512);

  // rope + assemble q/k, transpose v
  assemble_q<<<BB * HH * SS, 192, 0, stream>>>(qc_f, qr_f, cosT, sinT, qb);
  assemble_k<<<BB * HH * SS, 192, 0, stream>>>(kc_f, kr_f, cosT, sinT, kb);
  transpose_v<<<cdiv((size_t)BB * HH * SS * 128, 256), 256, 0, stream>>>(v_f, vtb);

  // flash attention
  mla_attn<<<dim3(SS / 128, HH, BB), 256, 0, stream>>>(qb, kb, vtb, aob);

  // output projection -> f32 result
  gemm_bf16<<<dim3(16, 32), 256, 0, stream>>>(aob, wot, out, 4096, 2048, 2048);
}
